// ActorNetwork_51324859187752
// MI455X (gfx1250) — compile-verified, matrix operand order checked by reading, not run on hardware
//
#include <hip/hip_runtime.h>

// ---------- types ----------
typedef __attribute__((ext_vector_type(16))) __bf16 v16bf;
typedef __attribute__((ext_vector_type(8)))  float  v8f;
typedef __attribute__((ext_vector_type(8)))  unsigned short v8us;

__device__ __forceinline__ unsigned short f2bf_bits(float f) {
    unsigned u = __float_as_uint(f);
    unsigned r = u + 0x7FFFu + ((u >> 16) & 1u);   // round-to-nearest-even
    return (unsigned short)(r >> 16);
}
__device__ __forceinline__ float bf2f(unsigned short h) {
    return __uint_as_float(((unsigned)h) << 16);
}

// A-fragment (16x32 bf16): lane L holds row M=L%16; K = 16*(i/8) + 8*(L/16) + i%8
__device__ __forceinline__ v16bf load_a_frag(const unsigned short* row, int k0, int half) {
    union { v16bf v; v8us h[2]; } u;
    u.h[0] = *(const v8us*)(row + k0 + 8 * half);
    u.h[1] = *(const v8us*)(row + k0 + 16 + 8 * half);
    return u.v;
}
// B-fragment pre-packed: 32 contiguous bytes per lane
__device__ __forceinline__ v16bf load_b_frag(const unsigned short* p) {
    union { v16bf v; v8us h[2]; } u;
    u.h[0] = *(const v8us*)(p);
    u.h[1] = *(const v8us*)(p + 8);
    return u.v;
}
__device__ __forceinline__ v8f wmma_bf16(v16bf a, v16bf b, v8f c) {
    return __builtin_amdgcn_wmma_f32_16x16x32_bf16(false, a, false, b, (short)0, c, false, false);
}

// ---------- weight packing: w[N][Kin] fp32 -> bf16 WMMA B-fragment order ----------
// packed[(nt*KC+kc)*512 + lane*16 + i] = w[nt*16 + lane%16][kc*32 + 16*(i/8) + 8*(lane/16) + i%8]
__global__ void pack_w_kernel(const float* __restrict__ w, unsigned short* __restrict__ packed,
                              int Kin, int KC, int total) {
    int tid = blockIdx.x * blockDim.x + threadIdx.x;
    if (tid >= total) return;
    int tl = tid >> 9;
    int r = tid & 511;
    int lane = r >> 4;
    int i = r & 15;
    int nt = tl / KC;
    int kc = tl - nt * KC;
    int n = nt * 16 + (lane & 15);
    int k = kc * 32 + 16 * (i >> 3) + 8 * (lane >> 4) + (i & 7);
    float v = (k < Kin) ? w[(size_t)n * Kin + k] : 0.0f;
    packed[tid] = f2bf_bits(v);
}

// ---------- fused edge MLP + masked aggregation ----------
// one workgroup per (b, i): 64 edges j through 8->512->512->512, then agg over j
__global__ __launch_bounds__(256) void edge_kernel(
    const float* __restrict__ obs,                 // (B=32, D=4, K=64)
    const float* __restrict__ e1_w, const float* __restrict__ e1_b,
    const unsigned short* __restrict__ packE2, const float* __restrict__ e2_b,
    const unsigned short* __restrict__ packE3, const float* __restrict__ e3_b,
    float* __restrict__ agg)                       // (B, K, 512)
{
    __shared__ unsigned short hA[64 * 512];        // 64 KB
    __shared__ unsigned short hB[64 * 512];        // 64 KB
    __shared__ float oL[64 * 4];
    __shared__ float maskF[64];

    const int i = blockIdx.x;
    const int b = blockIdx.y;
    const int tid = threadIdx.x;

    if (tid < 64) {
        float a0 = obs[b * 256 + 0 * 64 + tid];
        float a1 = obs[b * 256 + 1 * 64 + tid];
        float a2 = obs[b * 256 + 2 * 64 + tid];
        float a3 = obs[b * 256 + 3 * 64 + tid];
        oL[tid * 4 + 0] = a0; oL[tid * 4 + 1] = a1;
        oL[tid * 4 + 2] = a2; oL[tid * 4 + 3] = a3;
        maskF[tid] = (fabsf(a0) + fabsf(a1) + fabsf(a2) + fabsf(a3)) != 0.0f ? 1.0f : 0.0f;
    }
    __syncthreads();

    const float oI0 = oL[i * 4 + 0], oI1 = oL[i * 4 + 1];
    const float oI2 = oL[i * 4 + 2], oI3 = oL[i * 4 + 3];

    // layer 1 (K=8) in VALU: 64x512 outputs
    for (int idx = tid; idx < 64 * 512; idx += 256) {
        int j = idx >> 9;
        int n = idx & 511;
        const float* wr = e1_w + n * 8;
        float v = e1_b[n]
            + wr[0] * oI0 + wr[1] * oI1 + wr[2] * oI2 + wr[3] * oI3
            + wr[4] * oL[j * 4 + 0] + wr[5] * oL[j * 4 + 1]
            + wr[6] * oL[j * 4 + 2] + wr[7] * oL[j * 4 + 3];
        hA[idx] = f2bf_bits(v > 0.0f ? v : 0.0f);
    }
    __syncthreads();

    const int wave = tid >> 5, lane = tid & 31;
    const int half = lane >> 4, colc = lane & 15;
    const int mt = wave & 3;
    const int nbase = (wave >> 2) * 16;
    const int mrow = mt * 16 + colc;

    // layer 2: hA -> hB  (M=64, N=512, K=512)
    for (int nt = nbase; nt < nbase + 16; ++nt) {
        v8f acc = {};
        #pragma unroll
        for (int kc = 0; kc < 16; ++kc) {
            v16bf a = load_a_frag(&hA[mrow * 512], kc * 32, half);
            v16bf bb = load_b_frag(packE2 + (((size_t)(nt * 16 + kc)) << 9) + lane * 16);
            acc = wmma_bf16(a, bb, acc);
        }
        float bias = e2_b[nt * 16 + colc];
        #pragma unroll
        for (int r = 0; r < 8; ++r) {
            float v = acc[r] + bias;
            hB[(mt * 16 + half * 8 + r) * 512 + nt * 16 + colc] = f2bf_bits(v > 0.0f ? v : 0.0f);
        }
    }
    __syncthreads();

    // layer 3: hB -> hA
    for (int nt = nbase; nt < nbase + 16; ++nt) {
        v8f acc = {};
        #pragma unroll
        for (int kc = 0; kc < 16; ++kc) {
            v16bf a = load_a_frag(&hB[mrow * 512], kc * 32, half);
            v16bf bb = load_b_frag(packE3 + (((size_t)(nt * 16 + kc)) << 9) + lane * 16);
            acc = wmma_bf16(a, bb, acc);
        }
        float bias = e3_b[nt * 16 + colc];
        #pragma unroll
        for (int r = 0; r < 8; ++r) {
            float v = acc[r] + bias;
            hA[(mt * 16 + half * 8 + r) * 512 + nt * 16 + colc] = f2bf_bits(v > 0.0f ? v : 0.0f);
        }
    }
    __syncthreads();

    // masked aggregation over j
    const float mI = maskF[i];
    float cntJ = 0.0f;
    for (int j = 0; j < 64; ++j) cntJ += maskF[j];
    const float denom = fmaxf(mI * cntJ, 1e-6f);
    for (int n = tid; n < 512; n += 256) {
        float s = 0.0f;
        for (int j = 0; j < 64; ++j) s += maskF[j] * bf2f(hA[j * 512 + n]);
        agg[((size_t)b * 64 + i) * 512 + n] = mI * s / denom;
    }
}

// ---------- LayerNorm + node MLP (WMMA) + pooling + st_feat ----------
__global__ __launch_bounds__(256) void node_kernel(
    const float* __restrict__ obs, const float* __restrict__ state,
    const float* __restrict__ agg,
    const float* __restrict__ ln_g, const float* __restrict__ ln_bv,
    const unsigned short* __restrict__ packN1, const float* __restrict__ n1_b,
    const unsigned short* __restrict__ packN2, const float* __restrict__ n2_b,
    const float* __restrict__ layer_w, const float* __restrict__ layer_b,
    float* __restrict__ x)                          // (B, 1024)
{
    __shared__ __align__(16) unsigned char smem[69632 + 65536];
    unsigned short* ni = (unsigned short*)smem;               // 64 x 544 bf16
    float* nodeOut = (float*)smem;                            // 64 x 256 f32 (overlaid, used after n1)
    unsigned short* hN = (unsigned short*)(smem + 69632);     // 64 x 512 bf16
    __shared__ float maskF[64];
    __shared__ float meanS[64], rstdS[64];

    const int b = blockIdx.x;
    const int tid = threadIdx.x;
    const float* aggB = agg + (size_t)b * 64 * 512;

    if (tid < 64) {
        float a0 = obs[b * 256 + 0 * 64 + tid];
        float a1 = obs[b * 256 + 1 * 64 + tid];
        float a2 = obs[b * 256 + 2 * 64 + tid];
        float a3 = obs[b * 256 + 3 * 64 + tid];
        maskF[tid] = (fabsf(a0) + fabsf(a1) + fabsf(a2) + fabsf(a3)) != 0.0f ? 1.0f : 0.0f;
        float s = 0.0f, s2 = 0.0f;
        const float* ar = aggB + tid * 512;
        for (int n = 0; n < 512; ++n) { float v = ar[n]; s += v; s2 += v * v; }
        float mean = s * (1.0f / 512.0f);
        float var = s2 * (1.0f / 512.0f) - mean * mean;
        meanS[tid] = mean;
        rstdS[tid] = rsqrtf(fmaxf(var, 0.0f) + 1e-5f);
    }
    __syncthreads();

    // node_input: [o(4) | LN(agg)(512) | state(4) | pad(24)] -> 544 (17 K-chunks)
    for (int idx = tid; idx < 64 * 544; idx += 256) {
        int r = idx / 544;
        int c = idx - r * 544;
        float v;
        if (c < 4)            v = obs[b * 256 + c * 64 + r];
        else if (c < 516) { int n = c - 4;
                            v = (aggB[r * 512 + n] - meanS[r]) * rstdS[r] * ln_g[n] + ln_bv[n]; }
        else if (c < 520)     v = state[b * 4 + (c - 516)];
        else                  v = 0.0f;
        ni[idx] = f2bf_bits(v);
    }
    __syncthreads();

    const int wave = tid >> 5, lane = tid & 31;
    const int half = lane >> 4, colc = lane & 15;
    const int mt = wave & 3;
    const int mrow = mt * 16 + colc;

    // n1: 544 -> 512, relu
    {
        const int nbase = (wave >> 2) * 16;
        for (int nt = nbase; nt < nbase + 16; ++nt) {
            v8f acc = {};
            #pragma unroll
            for (int kc = 0; kc < 17; ++kc) {
                v16bf a = load_a_frag(&ni[mrow * 544], kc * 32, half);
                v16bf bb = load_b_frag(packN1 + (((size_t)(nt * 17 + kc)) << 9) + lane * 16);
                acc = wmma_bf16(a, bb, acc);
            }
            float bias = n1_b[nt * 16 + colc];
            #pragma unroll
            for (int r = 0; r < 8; ++r) {
                float v = acc[r] + bias;
                hN[(mt * 16 + half * 8 + r) * 512 + nt * 16 + colc] = f2bf_bits(v > 0.0f ? v : 0.0f);
            }
        }
    }
    __syncthreads();

    // n2: 512 -> 256 (no relu)
    {
        const int nbase = (wave >> 2) * 8;
        for (int nt = nbase; nt < nbase + 8; ++nt) {
            v8f acc = {};
            #pragma unroll
            for (int kc = 0; kc < 16; ++kc) {
                v16bf a = load_a_frag(&hN[mrow * 512], kc * 32, half);
                v16bf bb = load_b_frag(packN2 + (((size_t)(nt * 16 + kc)) << 9) + lane * 16);
                acc = wmma_bf16(a, bb, acc);
            }
            float bias = n2_b[nt * 16 + colc];
            #pragma unroll
            for (int r = 0; r < 8; ++r)
                nodeOut[(mt * 16 + half * 8 + r) * 256 + nt * 16 + colc] = acc[r] + bias;
        }
    }
    __syncthreads();

    // masked mean/max pooling -> x[512..1023]
    float cnt = 0.0f;
    for (int j = 0; j < 64; ++j) cnt += maskF[j];
    cnt = fmaxf(cnt, 1e-6f);
    {
        int n = tid;  // 256 threads, 256 columns
        float s = 0.0f, mx = -1e9f;
        for (int r = 0; r < 64; ++r) {
            float v = nodeOut[r * 256 + n];
            s += maskF[r] * v;
            mx = fmaxf(mx, maskF[r] != 0.0f ? v : -1e9f);
        }
        x[(size_t)b * 1024 + 512 + n] = s / cnt;
        x[(size_t)b * 1024 + 768 + n] = mx;
    }
    // st_feat -> x[0..511]
    for (int n = tid; n < 512; n += 256) {
        const float* wr = layer_w + n * 4;
        float v = layer_b[n]
            + wr[0] * state[b * 4 + 0] + wr[1] * state[b * 4 + 1]
            + wr[2] * state[b * 4 + 2] + wr[3] * state[b * 4 + 3];
        x[(size_t)b * 1024 + n] = v > 0.0f ? v : 0.0f;
    }
}

// ---------- mu / std heads (tiny, VALU) ----------
__global__ __launch_bounds__(256) void head_kernel(
    const float* __restrict__ x,
    const float* __restrict__ mu1_w, const float* __restrict__ mu1_b,
    const float* __restrict__ mu2_w, const float* __restrict__ mu2_b,
    const float* __restrict__ mu3_w, const float* __restrict__ mu3_b,
    const float* __restrict__ s1_w, const float* __restrict__ s1_b,
    const float* __restrict__ s2_w, const float* __restrict__ s2_b,
    const float* __restrict__ s3_w, const float* __restrict__ s3_b,
    float* __restrict__ out)   // [0..63] mu, [64..127] std
{
    __shared__ float xr[1024];
    __shared__ float h1m[256], h1s[256];
    __shared__ float h2m[128], h2s[128];
    const int b = blockIdx.x;
    const int tid = threadIdx.x;
    for (int k = tid; k < 1024; k += 256) xr[k] = x[(size_t)b * 1024 + k];
    __syncthreads();
    {
        int n = tid;
        float sm = mu1_b[n], ss = s1_b[n];
        const float* wm = mu1_w + (size_t)n * 1024;
        const float* ws = s1_w + (size_t)n * 1024;
        for (int k = 0; k < 1024; ++k) { sm += wm[k] * xr[k]; ss += ws[k] * xr[k]; }
        h1m[n] = sm > 0.0f ? sm : 0.0f;
        h1s[n] = ss > 0.0f ? ss : 0.0f;
    }
    __syncthreads();
    if (tid < 128) {
        int n = tid;
        float sm = mu2_b[n], ss = s2_b[n];
        const float* wm = mu2_w + n * 256;
        const float* ws = s2_w + n * 256;
        for (int k = 0; k < 256; ++k) { sm += wm[k] * h1m[k]; ss += ws[k] * h1s[k]; }
        h2m[n] = sm > 0.0f ? sm : 0.0f;
        h2s[n] = ss > 0.0f ? ss : 0.0f;
    }
    __syncthreads();
    if (tid < 2) {
        int a = tid;
        float sm = mu3_b[a], ss = s3_b[a];
        const float* wm = mu3_w + a * 128;
        const float* ws = s3_w + a * 128;
        for (int k = 0; k < 128; ++k) { sm += wm[k] * h2m[k]; ss += ws[k] * h2s[k]; }
        out[b * 2 + a] = sm;
        float sp = ss > 20.0f ? ss : log1pf(expf(ss));
        out[64 + b * 2 + a] = fminf(fmaxf(sp + 0.001f, 0.1f), 2.0f);
    }
}

extern "C" void kernel_launch(void* const* d_in, const int* in_sizes, int n_in,
                              void* d_out, int out_size, void* d_ws, size_t ws_size,
                              hipStream_t stream) {
    (void)in_sizes; (void)n_in; (void)out_size; (void)ws_size;
    const float* state   = (const float*)d_in[0];
    const float* obs     = (const float*)d_in[1];
    const float* layer_w = (const float*)d_in[2];
    const float* layer_b = (const float*)d_in[3];
    const float* e1_w    = (const float*)d_in[4];
    const float* e1_b    = (const float*)d_in[5];
    const float* e2_w    = (const float*)d_in[6];
    const float* e2_b    = (const float*)d_in[7];
    const float* e3_w    = (const float*)d_in[8];
    const float* e3_b    = (const float*)d_in[9];
    const float* ln_g    = (const float*)d_in[10];
    const float* ln_bv   = (const float*)d_in[11];
    const float* n1_w    = (const float*)d_in[12];
    const float* n1_b    = (const float*)d_in[13];
    const float* n2_w    = (const float*)d_in[14];
    const float* n2_b    = (const float*)d_in[15];
    const float* mu1_w   = (const float*)d_in[16];
    const float* mu1_b   = (const float*)d_in[17];
    const float* mu2_w   = (const float*)d_in[18];
    const float* mu2_b   = (const float*)d_in[19];
    const float* mu3_w   = (const float*)d_in[20];
    const float* mu3_b   = (const float*)d_in[21];
    const float* s1_w    = (const float*)d_in[22];
    const float* s1_b    = (const float*)d_in[23];
    const float* s2_w    = (const float*)d_in[24];
    const float* s2_b    = (const float*)d_in[25];
    const float* s3_w    = (const float*)d_in[26];
    const float* s3_b    = (const float*)d_in[27];
    float* out = (float*)d_out;

    char* ws = (char*)d_ws;
    size_t off = 0;
    unsigned short* packE2 = (unsigned short*)(ws + off); off += (size_t)512 * 512 * 2;
    unsigned short* packE3 = (unsigned short*)(ws + off); off += (size_t)512 * 512 * 2;
    unsigned short* packN1 = (unsigned short*)(ws + off); off += (size_t)544 * 512 * 2;
    unsigned short* packN2 = (unsigned short*)(ws + off); off += (size_t)512 * 256 * 2;
    float* aggW = (float*)(ws + off); off += (size_t)32 * 64 * 512 * 4;
    float* xW   = (float*)(ws + off); off += (size_t)32 * 1024 * 4;

    pack_w_kernel<<<(32 * 16 * 512) / 256, 256, 0, stream>>>(e2_w, packE2, 512, 16, 32 * 16 * 512);
    pack_w_kernel<<<(32 * 16 * 512) / 256, 256, 0, stream>>>(e3_w, packE3, 512, 16, 32 * 16 * 512);
    pack_w_kernel<<<(32 * 17 * 512) / 256, 256, 0, stream>>>(n1_w, packN1, 520, 17, 32 * 17 * 512);
    pack_w_kernel<<<(16 * 16 * 512) / 256, 256, 0, stream>>>(n2_w, packN2, 512, 16, 16 * 16 * 512);

    edge_kernel<<<dim3(64, 32), 256, 0, stream>>>(obs, e1_w, e1_b, packE2, e2_b, packE3, e3_b, aggW);
    node_kernel<<<32, 256, 0, stream>>>(obs, state, aggW, ln_g, ln_bv,
                                        packN1, n1_b, packN2, n2_b, layer_w, layer_b, xW);
    head_kernel<<<32, 256, 0, stream>>>(xW, mu1_w, mu1_b, mu2_w, mu2_b, mu3_w, mu3_b,
                                        s1_w, s1_b, s2_w, s2_b, s3_w, s3_b, out);
}
